// MyKAConvLSTM2_73065983640224
// MI455X (gfx1250) — matrix operand order repaired and checked by reading, not compiled, not run on hardware
//
#include <hip/hip_runtime.h>
#include <hip/hip_bf16.h>
#include <math.h>

// ---------------------------------------------------------------------------
// MI455X (gfx1250). Compute-bound model (~750 GF vs ~MBs of traffic), so all
// GEMM-shaped work runs on v_wmma_f32_16x16x32_f16 (f32 accumulate).
// Dominant support@t GEMMs additionally use the Tensor Data Mover to stream
// f16 A tiles into LDS (TENSORcnt + s_wait_tensorcnt), double-buffered.
// ---------------------------------------------------------------------------

typedef __attribute__((ext_vector_type(16))) _Float16 v16h;
typedef __attribute__((ext_vector_type(8)))  float    v8f;
typedef unsigned int u32x4 __attribute__((ext_vector_type(4)));
typedef int i32x4 __attribute__((ext_vector_type(4)));
typedef int i32x8 __attribute__((ext_vector_type(8)));

#define TPB 256
#define BM  128
#define BN  64
#define KC  32

#if defined(__has_builtin)
#  if __has_builtin(__builtin_amdgcn_tensor_load_to_lds)
#    define HAVE_TDM 1
#  endif
#endif
#ifndef HAVE_TDM
#  define HAVE_TDM 0
#endif

union FragH { unsigned u[8]; v16h v; };
union H8    { _Float16 h[8]; unsigned u[4]; };

__device__ __forceinline__ float activate(float v, int act) {
  if (act == 1) return v > 0.f ? v : 0.f;
  if (act == 2) return 1.f / (1.f + __expf(-v));
  if (act == 3) return tanhf(v);
  return v;
}

#if HAVE_TDM
__device__ __forceinline__ unsigned lds_addr_of(const void* p) {
  return (unsigned)(unsigned long long)(const __attribute__((address_space(3))) char*)p;
}
#endif

// ---------------------------------------------------------------------------
// Generic GEMM: Out = act(alpha*A@B + beta*Cin + bias), batched via blockIdx.z
// strides. K % 32 == 0 (true for all uses: 64/128/512/576/1024/2048).
// 128x64 tile, 8 wave32s, each wave owns a 32x32 quadrant (4 WMMAs/chunk),
// LDS double-buffered, branch-free clamped staging loads.
// ---------------------------------------------------------------------------
__global__ __launch_bounds__(TPB)
void k_gemm(const float* __restrict__ A,  long long sA, int lda,
            const float* __restrict__ Bm, long long sB, int ldb,
            const float* __restrict__ Cin, long long sC, int ldc,
            float* __restrict__ Out, long long sO, int ldo,
            const float* __restrict__ bias,
            int M, int Ncol, int Kd, float alpha, float beta, int act)
{
  __shared__ __align__(16) _Float16 As[2][BM * KC];   // [m][k]
  __shared__ __align__(16) _Float16 Bs[2][BN * KC];   // [n][k] transposed
  const int tid  = threadIdx.x;
  const int lane = tid & 31;
  const int wave = tid >> 5;
  const long long zb = blockIdx.z;
  A += zb * sA;  Bm += zb * sB;  Out += zb * sO;
  if (Cin) Cin += zb * sC;
  const int m0 = blockIdx.y * BM;
  const int n0 = blockIdx.x * BN;

  // A staging: thread covers half a row (16 f32). Row index is clamped so the
  // vector loads stay in-bounds and control flow stays uniform.
  const int ar_r = tid >> 1;
  const int ar_h = (tid & 1) << 4;
  const int agm  = m0 + ar_r;
  const long long arow = (agm < M) ? agm : (M - 1);
  const bool a_oob = (agm >= M);
  // B staging: thread covers 8 consecutive columns of one K row.
  const int bkk = tid >> 3;
  const int bnb = (tid & 7) << 3;
  const bool nfull = (n0 + BN) <= Ncol;

  float ar[16]; float br[8];

  auto stageA = [&](int k0) {
    const float* ap = A + arow * lda + k0 + ar_h;
#pragma unroll
    for (int j = 0; j < 4; ++j) {
      float4 q = ((const float4*)ap)[j];
      ar[4*j+0]=q.x; ar[4*j+1]=q.y; ar[4*j+2]=q.z; ar[4*j+3]=q.w;
    }
    if (a_oob) {
#pragma unroll
      for (int j = 0; j < 16; ++j) ar[j] = 0.f;
    }
  };
  auto stageB = [&](int k0) {
    const long long ro = (long long)(k0 + bkk) * ldb;
    if (nfull) {
      const float4* bp = (const float4*)(Bm + ro + n0 + bnb);
#pragma unroll
      for (int j = 0; j < 2; ++j) {
        float4 q = bp[j];
        br[4*j+0]=q.x; br[4*j+1]=q.y; br[4*j+2]=q.z; br[4*j+3]=q.w;
      }
    } else {            // Ncol tail (only the 64x12 head GEMM)
#pragma unroll
      for (int j = 0; j < 8; ++j) {
        int gn = n0 + bnb + j;
        br[j] = Bm[ro + (gn < Ncol ? gn : Ncol - 1)];
      }
#pragma unroll
      for (int j = 0; j < 8; ++j) if (n0 + bnb + j >= Ncol) br[j] = 0.f;
    }
  };
  auto storeA = [&](int buf) {
    H8 p0, p1;
#pragma unroll
    for (int j = 0; j < 8; ++j) { p0.h[j] = (_Float16)ar[j]; p1.h[j] = (_Float16)ar[8+j]; }
    unsigned* d = (unsigned*)&As[buf][ar_r * KC + ar_h];
#pragma unroll
    for (int j = 0; j < 4; ++j) d[j] = p0.u[j];
#pragma unroll
    for (int j = 0; j < 4; ++j) d[4+j] = p1.u[j];
  };
  auto storeB = [&](int buf) {
#pragma unroll
    for (int j = 0; j < 8; ++j) Bs[buf][(bnb + j) * KC + bkk] = (_Float16)br[j];
  };

  const int rq = wave & 3;   // 32-row quadrant
  const int cq = wave >> 2;  // 32-col quadrant
  v8f acc00 = {}, acc01 = {}, acc10 = {}, acc11 = {};

  stageA(0); stageB(0); storeA(0); storeB(0);
  __syncthreads();

  for (int k0 = 0; k0 < Kd; k0 += KC) {
    const int  cur = (k0 >> 5) & 1;
    const bool nx  = (k0 + KC) < Kd;
    if (nx) { stageA(k0 + KC); stageB(k0 + KC); }       // overlap with math
    if (lane == 0 && k0 + 2 * KC < Kd)
      __builtin_prefetch(A + arow * lda + k0 + 2 * KC, 0, 1);

    const unsigned* Au = (const unsigned*)As[cur];
    const unsigned* Bu = (const unsigned*)Bs[cur];
    const int ah  = (lane >> 4) * 4;   // K-half select (A)
    const int bh  = (lane >> 4) * 4;   // K-half select (B)
    const int l15 = lane & 15;
    FragH fa0, fa1, fb0, fb1;
    const int r0 = (rq * 32 + l15) * 16, r1 = (rq * 32 + 16 + l15) * 16;
    const int c0 = (cq * 32 + l15) * 16, c1 = (cq * 32 + 16 + l15) * 16;
#pragma unroll
    for (int i = 0; i < 4; ++i) {
      fa0.u[i] = Au[r0 + ah + i];  fa0.u[4+i] = Au[r0 + ah + 8 + i];
      fa1.u[i] = Au[r1 + ah + i];  fa1.u[4+i] = Au[r1 + ah + 8 + i];
    }
#pragma unroll
    for (int j = 0; j < 8; ++j) { fb0.u[j] = Bu[c0 + bh + j + (j & 4)]; fb1.u[j] = Bu[c1 + bh + j + (j & 4)]; }

    acc00 = __builtin_amdgcn_wmma_f32_16x16x32_f16(false, fa0.v, false, fb0.v, (short)0, acc00, false, false);
    acc01 = __builtin_amdgcn_wmma_f32_16x16x32_f16(false, fa0.v, false, fb1.v, (short)0, acc01, false, false);
    acc10 = __builtin_amdgcn_wmma_f32_16x16x32_f16(false, fa1.v, false, fb0.v, (short)0, acc10, false, false);
    acc11 = __builtin_amdgcn_wmma_f32_16x16x32_f16(false, fa1.v, false, fb1.v, (short)0, acc11, false, false);

    if (nx) { storeA(cur ^ 1); storeB(cur ^ 1); }
    __syncthreads();
  }

#pragma unroll
  for (int s = 0; s < 2; ++s) {
#pragma unroll
    for (int t = 0; t < 2; ++t) {
      v8f acc = s ? (t ? acc11 : acc10) : (t ? acc01 : acc00);
      const int gn = n0 + cq * 32 + t * 16 + (lane & 15);
      if (gn >= Ncol) continue;
#pragma unroll
      for (int i = 0; i < 8; ++i) {
        const int gm = m0 + rq * 32 + s * 16 + ((lane >> 4) << 3) + i;
        if (gm >= M) continue;
        float v = alpha * acc[i];
        if (Cin)  v += beta * Cin[(long long)gm * ldc + gn];
        if (bias) v += bias[gn];
        Out[(long long)gm * ldo + gn] = activate(v, act);
      }
    }
  }
}

// ---------------------------------------------------------------------------
// Support GEMM: A is a pre-converted f16 matrix (2048x2048). A tiles are DMA'd
// into LDS by the Tensor Data Mover; B (f32 feats) staged as in k_gemm.
// Full tiles only (M%128==0, Ncol%64==0, K%32==0).
// ---------------------------------------------------------------------------
__global__ __launch_bounds__(TPB)
void k_gemm_sup(const _Float16* __restrict__ Ah, int lda,
                const float* __restrict__ Bm, long long sB, int ldb,
                const float* __restrict__ Cin, long long sC, int ldc,
                float* __restrict__ Out, long long sO, int ldo,
                int M, int Ncol, int Kd, float alpha, float beta)
{
  __shared__ __align__(16) _Float16 As[2][BM * KC];
  __shared__ __align__(16) _Float16 Bs[2][BN * KC];
  const int tid  = threadIdx.x;
  const int lane = tid & 31;
  const int wave = tid >> 5;
  const long long zb = blockIdx.z;
  Bm += zb * sB;  Out += zb * sO;
  if (Cin) Cin += zb * sC;
  const int m0 = blockIdx.y * BM;
  const int n0 = blockIdx.x * BN;

  const int bkk = tid >> 3;
  const int bnb = (tid & 7) << 3;
  float br[8];

  auto stageB = [&](int k0) {
    const float4* bp = (const float4*)(Bm + (long long)(k0 + bkk) * ldb + n0 + bnb);
#pragma unroll
    for (int j = 0; j < 2; ++j) {
      float4 q = bp[j];
      br[4*j+0]=q.x; br[4*j+1]=q.y; br[4*j+2]=q.z; br[4*j+3]=q.w;
    }
  };
  auto storeB = [&](int buf) {
#pragma unroll
    for (int j = 0; j < 8; ++j) Bs[buf][(bnb + j) * KC + bkk] = (_Float16)br[j];
  };

#if HAVE_TDM
  // Issue a TDM 2D-tile load (tile 32 x 128 of 2-byte elements) into LDS.
  // D# per CDNA5 ISA 8.3/8.4: group0 = {count=1 | lds_addr | global_addr |
  // type=2}; group1 = {data_size=2B, tensor_dim0=K, tensor_dim1=M,
  // tile_dim0=32, tile_dim1=128, tensor_dim0_stride=lda}. Groups 2/3 unused.
  auto tdmA = [&](int k0, int buf) {
    if (wave == 0) {
      unsigned long long ga =
          (unsigned long long)(const void*)(Ah + (long long)m0 * lda + k0);
      u32x4 g0;
      g0[0] = 1u;                                  // count=1, user descriptor
      g0[1] = lds_addr_of(&As[buf][0]);            // lds_addr
      g0[2] = (unsigned)ga;                        // global_addr lo
      g0[3] = (unsigned)((ga >> 32) & 0x1ffffffull) | 0x80000000u; // hi | type=2
      i32x8 g1;
      g1[0] = 0x00010000;                          // data_size = 2 bytes
      g1[1] = (int)((unsigned)(Kd & 0xffff) << 16);                 // td0 lo16
      g1[2] = (int)(((unsigned)Kd >> 16) | ((unsigned)(M & 0xffff) << 16)); // td0 hi|td1 lo
      g1[3] = (int)(((unsigned)M >> 16) | ((unsigned)KC << 16));    // td1 hi|tile_dim0
      g1[4] = BM;                                  // tile_dim1
      g1[5] = lda;                                 // tensor_dim0_stride lo
      g1[6] = 0; g1[7] = 0;
      i32x4 z4 = (i32x4)0;
#if defined(__clang_major__) && (__clang_major__ >= 23)
      __builtin_amdgcn_tensor_load_to_lds(g0, g1, z4, z4, (i32x8)0, 0);
#else
      __builtin_amdgcn_tensor_load_to_lds(g0, g1, z4, z4, 0);
#endif
    }
  };
  auto tdmWait = [&]() {
    if (wave == 0) __builtin_amdgcn_s_wait_tensorcnt(0);
  };
#else
  // Fallback: plain vectorized f16 loads (no conversion needed).
  const int fr_r = tid >> 1;
  const int fr_h = (tid & 1) << 4;
  auto tdmA = [&](int k0, int buf) {
    const unsigned* ap =
        (const unsigned*)(Ah + (long long)(m0 + fr_r) * lda + k0 + fr_h);
    unsigned* d = (unsigned*)&As[buf][fr_r * KC + fr_h];
#pragma unroll
    for (int j = 0; j < 8; ++j) d[j] = ap[j];
  };
  auto tdmWait = [&]() {};
#endif

  const int rq = wave & 3;
  const int cq = wave >> 2;
  v8f acc00 = {}, acc01 = {}, acc10 = {}, acc11 = {};

  tdmA(0, 0);
  stageB(0); storeB(0);
  tdmWait();
  __syncthreads();

  for (int k0 = 0; k0 < Kd; k0 += KC) {
    const int  cur = (k0 >> 5) & 1;
    const bool nx  = (k0 + KC) < Kd;
    if (nx) { tdmA(k0 + KC, cur ^ 1); stageB(k0 + KC); }

    const unsigned* Au = (const unsigned*)As[cur];
    const unsigned* Bu = (const unsigned*)Bs[cur];
    const int ah  = (lane >> 4) * 4;
    const int bh  = (lane >> 4) * 4;
    const int l15 = lane & 15;
    FragH fa0, fa1, fb0, fb1;
    const int r0 = (rq * 32 + l15) * 16, r1 = (rq * 32 + 16 + l15) * 16;
    const int c0 = (cq * 32 + l15) * 16, c1 = (cq * 32 + 16 + l15) * 16;
#pragma unroll
    for (int i = 0; i < 4; ++i) {
      fa0.u[i] = Au[r0 + ah + i];  fa0.u[4+i] = Au[r0 + ah + 8 + i];
      fa1.u[i] = Au[r1 + ah + i];  fa1.u[4+i] = Au[r1 + ah + 8 + i];
    }
#pragma unroll
    for (int j = 0; j < 8; ++j) { fb0.u[j] = Bu[c0 + bh + j + (j & 4)]; fb1.u[j] = Bu[c1 + bh + j + (j & 4)]; }

    acc00 = __builtin_amdgcn_wmma_f32_16x16x32_f16(false, fa0.v, false, fb0.v, (short)0, acc00, false, false);
    acc01 = __builtin_amdgcn_wmma_f32_16x16x32_f16(false, fa0.v, false, fb1.v, (short)0, acc01, false, false);
    acc10 = __builtin_amdgcn_wmma_f32_16x16x32_f16(false, fa1.v, false, fb0.v, (short)0, acc10, false, false);
    acc11 = __builtin_amdgcn_wmma_f32_16x16x32_f16(false, fa1.v, false, fb1.v, (short)0, acc11, false, false);

    if (nx) { storeB(cur ^ 1); tdmWait(); }
    __syncthreads();
  }

#pragma unroll
  for (int s = 0; s < 2; ++s) {
#pragma unroll
    for (int t = 0; t < 2; ++t) {
      v8f acc = s ? (t ? acc11 : acc10) : (t ? acc01 : acc00);
      const int gn = n0 + cq * 32 + t * 16 + (lane & 15);
#pragma unroll
      for (int i = 0; i < 8; ++i) {
        const int gm = m0 + rq * 32 + s * 16 + ((lane >> 4) << 3) + i;
        float v = alpha * acc[i];
        if (Cin) v += beta * Cin[(long long)gm * ldc + gn];
        Out[(long long)gm * ldo + gn] = v;
      }
    }
  }
}

// ------------------------------- pointwise ---------------------------------

__global__ void k_f32_to_f16(const float* __restrict__ s, _Float16* __restrict__ d,
                             long long n) {
  long long i = (long long)blockIdx.x * blockDim.x + threadIdx.x;
  if (i < n) d[i] = (_Float16)s[i];
}

__global__ void k_zin1(const float* __restrict__ Z, const float* __restrict__ w1,
                       const float* __restrict__ b1, float* __restrict__ out,
                       long long total) {
  long long i = (long long)blockIdx.x * blockDim.x + threadIdx.x;
  if (i >= total) return;
  int d = i & 63; long long rc = i >> 6;
  float v = Z[rc * 2 + 0] * w1[d] + Z[rc * 2 + 1] * w1[64 + d] + b1[d];
  out[i] = v > 0.f ? v : 0.f;
}

__global__ void k_xin1(const float* __restrict__ X, const float* __restrict__ w1,
                       const float* __restrict__ b1, float* __restrict__ out,
                       long long total) {
  long long i = (long long)blockIdx.x * blockDim.x + threadIdx.x;
  if (i >= total) return;
  int d = i & 63; long long rc = i >> 6;
  float v = X[rc] * w1[d] + b1[d];
  out[i] = v > 0.f ? v : 0.f;
}

__global__ void k_te(const int* __restrict__ TE, const float* __restrict__ w1,
                     const float* __restrict__ b1, float* __restrict__ out,
                     int P, int Ttot, int total) {
  int i = blockIdx.x * blockDim.x + threadIdx.x;
  if (i >= total) return;
  int d = i & 63; int bp = i >> 6;
  int b = bp / P, p = bp - b * P;
  const int* te = TE + ((long long)b * Ttot + p) * 2;
  float v = w1[te[0] * 64 + d] + w1[(7 + te[1]) * 64 + d] + b1[d];
  out[i] = v > 0.f ? v : 0.f;
}

__global__ void k_add_ste(float* __restrict__ x, const float* __restrict__ se,
                          const float* __restrict__ te2,
                          const float* __restrict__ extra,
                          int R, long long total) {
  long long i = (long long)blockIdx.x * blockDim.x + threadIdx.x;
  if (i >= total) return;
  int d = i & 63; long long rr = i >> 6;
  int r = (int)(rr % R); long long bp = rr / R;
  float v = x[i] + se[r * 64 + d] + te2[bp * 64 + d];
  if (extra) v += extra[i];
  x[i] = v;
}

__global__ void k_im2col(const float* __restrict__ src, long long bstride,
                         float* __restrict__ dst, long long total) {
  long long i = (long long)blockIdx.x * blockDim.x + threadIdx.x;
  if (i >= total) return;
  int e = (int)(i % 576); long long row = i / 576;
  int b = (int)(row >> 10); int hw = (int)(row & 1023);
  int hh = hw >> 5, ww = hw & 31;
  int tap = e >> 6, dch = e & 63;
  int kh = tap / 3, kw = tap - 3 * kh;
  int sh = hh + kh - 1, sw = ww + kw - 1;
  float v = 0.f;
  if (sh >= 0 && sh < 32 && sw >= 0 && sw < 32)
    v = src[(long long)b * bstride + ((sh << 5) + sw) * 64 + dch];
  dst[i] = v;
}

__device__ __forceinline__ float hsig(float x) {
  float v = 0.2f * x + 0.5f;
  return v < 0.f ? 0.f : (v > 1.f ? 1.f : v);
}

__global__ void k_lstm_cell(const float* __restrict__ g, float* __restrict__ c,
                            float* __restrict__ h, long long total) {
  long long i = (long long)blockIdx.x * blockDim.x + threadIdx.x;
  if (i >= total) return;
  int d = i & 63; long long rc = i >> 6;
  const float* gg = g + rc * 256;
  float iv = hsig(gg[d]), fv = hsig(gg[64 + d]);
  float cc = tanhf(gg[128 + d]), ov = hsig(gg[192 + d]);
  float cn = fv * c[i] + iv * cc;
  c[i] = cn;
  h[i] = ov * tanhf(cn);
}

__global__ void k_slice0(const float* __restrict__ xt, long long xbstride,
                         const float* __restrict__ h, const float* __restrict__ ru,
                         float* __restrict__ feats, int Nn, long long total) {
  long long i = (long long)blockIdx.x * blockDim.x + threadIdx.x;
  if (i >= total) return;
  int d = i & 63; long long bn = i >> 6;
  int b = (int)(bn / Nn); int n = (int)(bn - (long long)b * Nn);
  float hv = h[i];
  if (ru) hv *= ru[bn * 128 + d];
  float* f = feats + bn * 512;
  f[d]      = xt[(long long)b * xbstride + (long long)n * 64 + d];
  f[64 + d] = hv;
}

__global__ void k_hupdate(float* __restrict__ h, const float* __restrict__ ru,
                          const float* __restrict__ c, long long total) {
  long long i = (long long)blockIdx.x * blockDim.x + threadIdx.x;
  if (i >= total) return;
  int d = i & 63; long long bn = i >> 6;
  float u = ru[bn * 128 + 64 + d];
  h[i] = u * h[i] + (1.f - u) * c[i];
}

__global__ void k_transpose_out(const float* __restrict__ y, float* __restrict__ out,
                                int Nn, int Qc, long long total) {
  long long i = (long long)blockIdx.x * blockDim.x + threadIdx.x;
  if (i >= total) return;
  int n = (int)(i % Nn); long long bq = i / Nn;
  int q = (int)(bq % Qc); long long b = bq / Qc;
  out[i] = y[(b * Nn + n) * Qc + q];
}

// ------------------------------- host side ---------------------------------

static inline void launch_gemm(hipStream_t s,
                               const float* A, long long sA, int lda,
                               const float* B, long long sB, int ldb,
                               const float* C, long long sC, int ldc,
                               float* O, long long sO, int ldo,
                               const float* bias, int M, int Ncol, int Kd,
                               float alpha, float beta, int act, int batch) {
  dim3 g((Ncol + BN - 1) / BN, (M + BM - 1) / BM, batch);
  k_gemm<<<g, dim3(TPB), 0, s>>>(A, sA, lda, B, sB, ldb, C, sC, ldc,
                                 O, sO, ldo, bias, M, Ncol, Kd, alpha, beta, act);
}

static inline void launch_sup(hipStream_t s, const _Float16* Ah, int lda,
                              const float* B, long long sB, int ldb,
                              const float* C, long long sC, int ldc,
                              float* O, long long sO, int ldo,
                              int M, int Ncol, int Kd,
                              float alpha, float beta, int batch) {
  dim3 g(Ncol / BN, M / BM, batch);
  k_gemm_sup<<<g, dim3(TPB), 0, s>>>(Ah, lda, B, sB, ldb, C, sC, ldc,
                                     O, sO, ldo, M, Ncol, Kd, alpha, beta);
}

static inline dim3 pw_grid(long long total) {
  return dim3((unsigned)((total + TPB - 1) / TPB));
}

extern "C" void kernel_launch(void* const* d_in, const int* in_sizes, int n_in,
                              void* d_out, int out_size, void* d_ws, size_t ws_size,
                              hipStream_t stream) {
  (void)in_sizes; (void)n_in; (void)out_size; (void)ws_size;
  const int Bc = 8, Pc = 12, Qc = 12, Nn = 2048, Cc = 1024, Ttot = 24;

  const float* X    = (const float*)d_in[0];
  const float* Z    = (const float*)d_in[1];
  const int*   TE   = (const int*)d_in[2];
  const float* SE_X = (const float*)d_in[3];
  const float* SE_Z = (const float*)d_in[4];
  const float* sx_se_w1 = (const float*)d_in[5];
  const float* sx_se_b1 = (const float*)d_in[6];
  const float* sx_se_w2 = (const float*)d_in[7];
  const float* sx_se_b2 = (const float*)d_in[8];
  const float* sx_te_w1 = (const float*)d_in[9];
  const float* sx_te_b1 = (const float*)d_in[10];
  const float* sx_te_w2 = (const float*)d_in[11];
  const float* sx_te_b2 = (const float*)d_in[12];
  const float* sz_se_w1 = (const float*)d_in[13];
  const float* sz_se_b1 = (const float*)d_in[14];
  const float* sz_se_w2 = (const float*)d_in[15];
  const float* sz_se_b2 = (const float*)d_in[16];
  const float* sz_te_w1 = (const float*)d_in[17];
  const float* sz_te_b1 = (const float*)d_in[18];
  const float* sz_te_w2 = (const float*)d_in[19];
  const float* sz_te_b2 = (const float*)d_in[20];
  const float* zin_w1  = (const float*)d_in[21];
  const float* zin_b1  = (const float*)d_in[22];
  const float* zin_w2  = (const float*)d_in[23];
  const float* zin_b2  = (const float*)d_in[24];
  const float* xin_w1  = (const float*)d_in[25];
  const float* xin_b1  = (const float*)d_in[26];
  const float* xin_w2  = (const float*)d_in[27];
  const float* xin_b2  = (const float*)d_in[28];
  const float* xout_w1 = (const float*)d_in[29];
  const float* xout_b1 = (const float*)d_in[30];
  const float* xout_w2 = (const float*)d_in[31];
  const float* xout_b2 = (const float*)d_in[32];
  const float* clstm_wx = (const float*)d_in[33];
  const float* clstm_wh = (const float*)d_in[34];
  const float* zdec_w  = (const float*)d_in[35];
  const float* zdec_b  = (const float*)d_in[36];
  const float* w_gate  = (const float*)d_in[37];
  const float* b_gate  = (const float*)d_in[38];
  const float* w_cand  = (const float*)d_in[39];
  const float* b_cand  = (const float*)d_in[40];
  const float* adj_gr  = (const float*)d_in[41];
  const float* support = (const float*)d_in[42];

  char* wp = (char*)d_ws;
  auto alloc = [&](long long nfloat) -> float* {
    float* p = (float*)wp; wp += nfloat * (long long)sizeof(float); return p;
  };
  float* tmpSE = alloc((long long)Nn * 64);
  float* seZ   = alloc((long long)Cc * 64);
  float* seX   = alloc((long long)Nn * 64);
  float* te1   = alloc(96 * 64);
  float* te2z  = alloc(96 * 64);
  float* te2x  = alloc(96 * 64);
  float* z1    = alloc((long long)Bc * Pc * Cc * 64);
  float* zbuf  = alloc((long long)Bc * Pc * Cc * 64);
  float* colb  = alloc((long long)8192 * 576);
  float* gates = alloc((long long)8192 * 256);
  float* hls   = alloc((long long)8192 * 64);
  float* cls   = alloc((long long)8192 * 64);
  float* zdec  = alloc((long long)8192 * 768);
  float* zx    = alloc((long long)Bc * Pc * Nn * 64);
  float* x1    = alloc((long long)Bc * Pc * Nn * 64);
  float* xbuf  = alloc((long long)Bc * Pc * Nn * 64);
  float* hN    = alloc((long long)Bc * Nn * 64);
  float* fg    = alloc((long long)Bc * Nn * 512);
  float* fc    = alloc((long long)Bc * Nn * 512);
  float* ru    = alloc((long long)Bc * Nn * 128);
  float* ccand = alloc((long long)Bc * Nn * 64);
  float* t1b   = alloc((long long)Bc * Nn * 64);
  float* ybuf  = alloc((long long)Bc * Nn * 12);
  _Float16* supH = (_Float16*)alloc((long long)Nn * Nn / 2);  // 8 MB f16 copy

  // ---- spatial embeddings
  launch_gemm(stream, SE_Z, 0, 64, sz_se_w1, 0, 64, nullptr, 0, 0,
              tmpSE, 0, 64, sz_se_b1, Cc, 64, 64, 1.f, 0.f, 1, 1);
  launch_gemm(stream, tmpSE, 0, 64, sz_se_w2, 0, 64, nullptr, 0, 0,
              seZ, 0, 64, sz_se_b2, Cc, 64, 64, 1.f, 0.f, 0, 1);
  launch_gemm(stream, SE_X, 0, 64, sx_se_w1, 0, 64, nullptr, 0, 0,
              tmpSE, 0, 64, sx_se_b1, Nn, 64, 64, 1.f, 0.f, 1, 1);
  launch_gemm(stream, tmpSE, 0, 64, sx_se_w2, 0, 64, nullptr, 0, 0,
              seX, 0, 64, sx_se_b2, Nn, 64, 64, 1.f, 0.f, 0, 1);

  // ---- temporal embeddings
  k_te<<<pw_grid(96 * 64), TPB, 0, stream>>>(TE, sz_te_w1, sz_te_b1, te1, Pc, Ttot, 96 * 64);
  launch_gemm(stream, te1, 0, 64, sz_te_w2, 0, 64, nullptr, 0, 0,
              te2z, 0, 64, sz_te_b2, 96, 64, 64, 1.f, 0.f, 0, 1);
  k_te<<<pw_grid(96 * 64), TPB, 0, stream>>>(TE, sx_te_w1, sx_te_b1, te1, Pc, Ttot, 96 * 64);
  launch_gemm(stream, te1, 0, 64, sx_te_w2, 0, 64, nullptr, 0, 0,
              te2x, 0, 64, sx_te_b2, 96, 64, 64, 1.f, 0.f, 0, 1);

  // ---- z input path
  long long zTot = (long long)Bc * Pc * Cc * 64;
  k_zin1<<<pw_grid(zTot), TPB, 0, stream>>>(Z, zin_w1, zin_b1, z1, zTot);
  launch_gemm(stream, z1, 0, 64, zin_w2, 0, 64, nullptr, 0, 0,
              zbuf, 0, 64, zin_b2, Bc * Pc * Cc, 64, 64, 1.f, 0.f, 0, 1);
  k_add_ste<<<pw_grid(zTot), TPB, 0, stream>>>(zbuf, seZ, te2z, nullptr, Cc, zTot);

  // ---- ConvLSTM (im2col + WMMA GEMM)
  hipMemsetAsync(hls, 0, (long long)8192 * 64 * sizeof(float), stream);
  hipMemsetAsync(cls, 0, (long long)8192 * 64 * sizeof(float), stream);
  const long long colTot = (long long)8192 * 576;
  for (int t = 0; t < Pc; ++t) {
    k_im2col<<<pw_grid(colTot), TPB, 0, stream>>>(
        zbuf + (long long)t * Cc * 64, (long long)Pc * Cc * 64, colb, colTot);
    launch_gemm(stream, colb, 0, 576, clstm_wx, 0, 256, nullptr, 0, 0,
                gates, 0, 256, nullptr, 8192, 256, 576, 1.f, 0.f, 0, 1);
    k_im2col<<<pw_grid(colTot), TPB, 0, stream>>>(
        hls, (long long)Cc * 64, colb, colTot);
    launch_gemm(stream, colb, 0, 576, clstm_wh, 0, 256, gates, 0, 256,
                gates, 0, 256, nullptr, 8192, 256, 576, 1.f, 1.f, 0, 1);
    k_lstm_cell<<<pw_grid((long long)8192 * 64), TPB, 0, stream>>>(
        gates, cls, hls, (long long)8192 * 64);
  }

  // ---- z decoder
  launch_gemm(stream, hls, 0, 64, zdec_w, 0, 768, nullptr, 0, 0,
              zdec, 0, 768, zdec_b, 8192, 768, 64, 1.f, 0.f, 1, 1);

  // ---- zx = adj_gr @ z  (batched over b, looped over p)
  for (int p = 0; p < Pc; ++p) {
    launch_gemm(stream, adj_gr, 0, Cc,
                zdec + (long long)p * 64, (long long)Cc * 768, 768,
                nullptr, 0, 0,
                zx + (long long)p * Nn * 64, (long long)Pc * Nn * 64, 64,
                nullptr, Nn, 64, Cc, 1.f, 0.f, 0, Bc);
  }

  // ---- x input path + assemble
  long long xTot = (long long)Bc * Pc * Nn * 64;
  k_xin1<<<pw_grid(xTot), TPB, 0, stream>>>(X, xin_w1, xin_b1, x1, xTot);
  launch_gemm(stream, x1, 0, 64, xin_w2, 0, 64, nullptr, 0, 0,
              xbuf, 0, 64, xin_b2, Bc * Pc * Nn, 64, 64, 1.f, 0.f, 0, 1);
  k_add_ste<<<pw_grid(xTot), TPB, 0, stream>>>(xbuf, seX, te2x, zx, Nn, xTot);

  // ---- DCGRU (dominant GEMMs on the TDM+WMMA path)
  k_f32_to_f16<<<pw_grid((long long)Nn * Nn), TPB, 0, stream>>>(
      support, supH, (long long)Nn * Nn);
  hipMemsetAsync(hN, 0, (long long)Bc * Nn * 64 * sizeof(float), stream);
  const long long bnTot = (long long)Bc * Nn * 64;
  const long long sF = (long long)Nn * 512;
  for (int p = 0; p < Pc; ++p) {
    const float* xt = xbuf + (long long)p * Nn * 64;
    const long long xbs = (long long)Pc * Nn * 64;

    k_slice0<<<pw_grid(bnTot), TPB, 0, stream>>>(xt, xbs, hN, nullptr, fg, Nn, bnTot);
    launch_sup(stream, supH, Nn, fg + 0, sF, 512, nullptr, 0, 0,
               fg + 128, sF, 512, Nn, 128, Nn, 1.f, 0.f, Bc);
    launch_sup(stream, supH, Nn, fg + 128, sF, 512, fg + 0, sF, 512,
               fg + 256, sF, 512, Nn, 128, Nn, 2.f, -1.f, Bc);
    launch_sup(stream, supH, Nn, fg + 256, sF, 512, fg + 128, sF, 512,
               fg + 384, sF, 512, Nn, 128, Nn, 2.f, -1.f, Bc);
    launch_gemm(stream, fg, 0, 512, w_gate, 0, 128, nullptr, 0, 0,
                ru, 0, 128, b_gate, Bc * Nn, 128, 512, 1.f, 0.f, 2, 1);

    k_slice0<<<pw_grid(bnTot), TPB, 0, stream>>>(xt, xbs, hN, ru, fc, Nn, bnTot);
    launch_sup(stream, supH, Nn, fc + 0, sF, 512, nullptr, 0, 0,
               fc + 128, sF, 512, Nn, 128, Nn, 1.f, 0.f, Bc);
    launch_sup(stream, supH, Nn, fc + 128, sF, 512, fc + 0, sF, 512,
               fc + 256, sF, 512, Nn, 128, Nn, 2.f, -1.f, Bc);
    launch_sup(stream, supH, Nn, fc + 256, sF, 512, fc + 128, sF, 512,
               fc + 384, sF, 512, Nn, 128, Nn, 2.f, -1.f, Bc);
    launch_gemm(stream, fc, 0, 512, w_cand, 0, 64, nullptr, 0, 0,
                ccand, 0, 64, b_cand, Bc * Nn, 64, 512, 1.f, 0.f, 3, 1);

    k_hupdate<<<pw_grid(bnTot), TPB, 0, stream>>>(hN, ru, ccand, bnTot);
  }

  // ---- output head
  launch_gemm(stream, hN, 0, 64, xout_w1, 0, 64, nullptr, 0, 0,
              t1b, 0, 64, xout_b1, Bc * Nn, 64, 64, 1.f, 0.f, 1, 1);
  launch_gemm(stream, t1b, 0, 64, xout_w2, 0, 12, nullptr, 0, 0,
              ybuf, 0, 12, xout_b2, Bc * Nn, 12, 64, 1.f, 0.f, 0, 1);
  long long oTot = (long long)Bc * Qc * Nn;
  k_transpose_out<<<pw_grid(oTot), TPB, 0, stream>>>(ybuf, (float*)d_out, Nn, Qc, oTot);
}
